// AgentAttention_65180423685315
// MI455X (gfx1250) — matrix operand order repaired and checked by reading, compiled, not run on hardware
//
#include <hip/hip_runtime.h>
#include <hip/hip_bf16.h>

typedef __attribute__((ext_vector_type(16))) _Float16 v16h;
typedef __attribute__((ext_vector_type(8)))  float    v8f;

#define BQ 16
#define NQ 3136
#define CQ 512
#define HH 56
#define HEADS 8
#define AGENT 49
#define DHEAD 64
#define MROWS (BQ * NQ)   // 50176
#define SCALE 0.125f      // 1/sqrt(64)

// ---------------------------------------------------------------------------
// WMMA fragment helpers (CDNA5 16x16x32 f16 layouts, wave32)
// A (16x32, MxK): lane r = lane&15, hi = lane>>4; half j -> k = (j&7)+8*hi+16*(j>>3)
// B (32x16, KxN): lane n = lane&15; half j -> k = (j&7)+8*hi+16*(j>>3)
// C/D (16x16 f32): VGPR g -> M = g+8*hi; N = lane&15
// ---------------------------------------------------------------------------

static __device__ __forceinline__ v16h frag_a_ptr(const _Float16* rowp, int hi) {
    v16h a;
    const _Float16* p = rowp + 8 * hi;
#pragma unroll
    for (int j = 0; j < 8; ++j) { a[j] = p[j]; a[j + 8] = p[16 + j]; }
    return a;
}

static __device__ __forceinline__ v16h frag_a_zero() {
    v16h a;
#pragma unroll
    for (int j = 0; j < 16; ++j) a[j] = (_Float16)0.0f;
    return a;
}

static __device__ __forceinline__ v16h frag_b_ptr(const _Float16* kp, int ld, int hi) {
    v16h b;
#pragma unroll
    for (int j = 0; j < 8; ++j) { b[j] = kp[(8 * hi + j) * ld]; b[j + 8] = kp[(16 + 8 * hi + j) * ld]; }
    return b;
}

#define WMMA(c, a, b) \
    __builtin_amdgcn_wmma_f32_16x16x32_f16(false, (a), false, (b), (short)0, (c), false, false)

// ---------------------------------------------------------------------------
// CDNA5 async global->LDS copies (ASYNCcnt-tracked DMA, bypasses VGPRs).
// Low 32 bits of a flat shared-memory address are the LDS byte offset
// (ISA 10.2). Async loads complete in order,
// so s_wait_asynccnt N drains all but the newest N copies.
// ---------------------------------------------------------------------------
static __device__ __forceinline__ void async_b128(void* lds, const void* gsrc) {
    unsigned ldsoff = (unsigned)(unsigned long long)lds;
    asm volatile("global_load_async_to_lds_b128 %0, %1, off"
                 :: "v"(ldsoff), "v"(gsrc) : "memory");
}
static __device__ __forceinline__ void async_wait0() {
    asm volatile("s_wait_asynccnt 0" ::: "memory");
}
static __device__ __forceinline__ void async_wait6() {
    asm volatile("s_wait_asynccnt 6" ::: "memory");
}

// ---------------------------------------------------------------------------
// 0) fp32 -> fp16 elementwise convert
// ---------------------------------------------------------------------------
__global__ void cvt_f32_f16(const float* __restrict__ src, _Float16* __restrict__ dst, long n) {
    long i = (long)blockIdx.x * blockDim.x + threadIdx.x;
    if (i < n) dst[i] = (_Float16)src[i];
}

// ---------------------------------------------------------------------------
// 1) WMMA GEMM: out[M x NN] = A16[M x KK] * B16[KK x NN] (+ bias)
//    block tile 128x256, 8 waves (4x2), wave tile 32x128 (2x8 of 16x16)
//    Double-buffered LDS tiles staged with GLOBAL_LOAD_ASYNC_TO_LDS_B128:
//    tile k+1 is DMA'd while tile k feeds the WMMA pipes.
// ---------------------------------------------------------------------------
template <int NN, int KK, bool HAS_BIAS, bool OUT_H>
__global__ __launch_bounds__(256) void gemm16(const _Float16* __restrict__ A,
                                              const _Float16* __restrict__ B16,
                                              const float* __restrict__ bias,
                                              float* __restrict__ outF,
                                              _Float16* __restrict__ outH) {
    __shared__ _Float16 As[2][128][40];    // 2 x (128 x 32) + pad
    __shared__ _Float16 Bs[2][32][264];    // 2 x (32 x 256) + pad
    const int t = threadIdx.x;
    const int wave = t >> 5, lane = t & 31, lr = lane & 15, hi = lane >> 4;
    const int wm = wave >> 1, wn = wave & 1;
    const int m0blk = blockIdx.y * 128;
    const int n0blk = blockIdx.x * 256;

    auto issue = [&](int k0, int buf) {
        // A tile: 128x32 halves = 512 16B chunks (2 per thread)
#pragma unroll
        for (int cidx = 0; cidx < 2; ++cidx) {
            const int c = t + cidx * 256;
            const int row = c >> 2, j = (c & 3) * 8;
            async_b128(&As[buf][row][j], A + (long)(m0blk + row) * KK + k0 + j);
        }
        // B tile: 32x256 halves = 1024 16B chunks (4 per thread)
#pragma unroll
        for (int cidx = 0; cidx < 4; ++cidx) {
            const int c = t + cidx * 256;
            const int row = c >> 5, j = (c & 31) * 8;
            async_b128(&Bs[buf][row][j], B16 + (long)(k0 + row) * NN + n0blk + j);
        }
    };

    v8f acc[2][8];
#pragma unroll
    for (int i = 0; i < 2; ++i)
#pragma unroll
        for (int j = 0; j < 8; ++j) acc[i][j] = (v8f){0.f, 0.f, 0.f, 0.f, 0.f, 0.f, 0.f, 0.f};

    issue(0, 0);
    int cur = 0;
    for (int k0 = 0; k0 < KK; k0 += 32) {
        if (k0 + 32 < KK) {           // prefetch next tile into other buffer
            issue(k0 + 32, cur ^ 1);
            async_wait6();            // drain the 6 copies of the current tile
        } else {
            async_wait0();
        }
        __syncthreads();

        v16h afr[2];
#pragma unroll
        for (int mt = 0; mt < 2; ++mt)
            afr[mt] = frag_a_ptr(&As[cur][wm * 32 + mt * 16 + lr][0], hi);
#pragma unroll
        for (int nt = 0; nt < 8; ++nt) {
            v16h bfr = frag_b_ptr(&Bs[cur][0][wn * 128 + nt * 16 + lr], 264, hi);
#pragma unroll
            for (int mt = 0; mt < 2; ++mt)
                acc[mt][nt] = WMMA(acc[mt][nt], afr[mt], bfr);
        }
        __syncthreads();
        cur ^= 1;
    }

#pragma unroll
    for (int mt = 0; mt < 2; ++mt)
#pragma unroll
        for (int nt = 0; nt < 8; ++nt) {
            const int gn = n0blk + wn * 128 + nt * 16 + lr;
            float badd = 0.f;
            if (HAS_BIAS) badd = bias[gn];
#pragma unroll
            for (int g = 0; g < 8; ++g) {
                const int gm = m0blk + wm * 32 + mt * 16 + g + 8 * hi;
                const float v = acc[mt][nt][g] + badd;
                if (OUT_H) outH[(long)gm * NN + gn] = (_Float16)v;
                else       outF[(long)gm * NN + gn] = v;
            }
        }
}

// ---------------------------------------------------------------------------
// 2) 8x8 mean pool of q -> agent tokens (b, 49, 512), f16
// ---------------------------------------------------------------------------
__global__ __launch_bounds__(256) void pool_agent(const _Float16* __restrict__ q16,
                                                  _Float16* __restrict__ ag) {
    const int a = blockIdx.x;   // 0..48
    const int b = blockIdx.y;   // 0..15
    const int ay = a / 7, ax = a % 7;
    for (int c = threadIdx.x; c < CQ; c += 256) {
        float s = 0.f;
#pragma unroll
        for (int dy = 0; dy < 8; ++dy)
#pragma unroll
            for (int dx = 0; dx < 8; ++dx) {
                const int n = (ay * 8 + dy) * HH + ax * 8 + dx;
                s += (float)q16[((long)(b * NQ + n)) * CQ + c];
            }
        ag[((long)(b * AGENT + a)) * CQ + c] = (_Float16)(s * (1.0f / 64.0f));
    }
}

// ---------------------------------------------------------------------------
// 3) Bias tables: bilinear 7x7 -> 56x56 (half-pixel centers, edge clamp)
// ---------------------------------------------------------------------------
__global__ void bias_build(const float* __restrict__ an, const float* __restrict__ na,
                           const float* __restrict__ ah, const float* __restrict__ aw,
                           const float* __restrict__ ha, const float* __restrict__ wa,
                           float* __restrict__ bak, float* __restrict__ bqa) {
    const int idx = blockIdx.x * 256 + threadIdx.x;
    if (idx >= HEADS * AGENT * NQ) return;
    const int n = idx % NQ;
    const int haid = idx / NQ;
    const int a = haid % AGENT, h = haid / AGENT;
    const int y = n / HH, x = n % HH;

    const float sy = (y + 0.5f) * 0.125f - 0.5f;
    const float sx = (x + 0.5f) * 0.125f - 0.5f;
    int iy0 = (int)floorf(sy); const float ty = sy - (float)iy0;
    int ix0 = (int)floorf(sx); const float tx = sx - (float)ix0;
    int iy1 = min(iy0 + 1, 6); iy0 = max(0, min(iy0, 6));
    int ix1 = min(ix0 + 1, 6); ix0 = max(0, min(ix0, 6));

    const long tb = (long)(h * AGENT + a) * 49;
    const float* tA = an + tb;
    const float* tN = na + tb;
    const float vA = (1.f - ty) * ((1.f - tx) * tA[iy0 * 7 + ix0] + tx * tA[iy0 * 7 + ix1]) +
                     ty * ((1.f - tx) * tA[iy1 * 7 + ix0] + tx * tA[iy1 * 7 + ix1]);
    const float vN = (1.f - ty) * ((1.f - tx) * tN[iy0 * 7 + ix0] + tx * tN[iy0 * 7 + ix1]) +
                     ty * ((1.f - tx) * tN[iy1 * 7 + ix0] + tx * tN[iy1 * 7 + ix1]);

    bak[(long)(h * AGENT + a) * NQ + n] = vA + ah[h * AGENT + a] + aw[h * AGENT + a];
    bqa[((long)(h * NQ + n)) * 64 + a] = vN + ha[h * AGENT + a] + wa[h * AGENT + a];
}

// ---------------------------------------------------------------------------
// 4) Stage-1 scores: S1[b,h,a,n] = scale * agent(49x64) . K^T + bias_ak
//    S1 padded to 64 agent rows per (b,h) -> unguarded stores (no exec churn)
// ---------------------------------------------------------------------------
__global__ __launch_bounds__(128) void s1_scores(const _Float16* __restrict__ ag,
                                                 const _Float16* __restrict__ kv,
                                                 const float* __restrict__ bak,
                                                 _Float16* __restrict__ S1) {
    const int b = blockIdx.z, h = blockIdx.y;
    const int n0 = blockIdx.x * 64;
    __shared__ _Float16 kt[64][72];  // [d][n]
    const int t = threadIdx.x;
    for (int e = t; e < 64 * 64; e += 128) {
        const int n = e >> 6, d = e & 63;
        kt[d][n] = kv[((long)(b * NQ + n0 + n)) * 1024 + h * DHEAD + d];
    }
    __syncthreads();

    const int wave = t >> 5, lane = t & 31, lr = lane & 15, hi = lane >> 4;
    const int m0 = wave * 16;
    const int arow = m0 + lr;
    const _Float16* abase = ag + ((long)(b * AGENT + min(arow, AGENT - 1))) * CQ + h * DHEAD;

    v8f acc[4];
#pragma unroll
    for (int i = 0; i < 4; ++i) acc[i] = (v8f){0.f, 0.f, 0.f, 0.f, 0.f, 0.f, 0.f, 0.f};

#pragma unroll
    for (int ks = 0; ks < 2; ++ks) {
        v16h a = frag_a_ptr(abase + ks * 32, hi);
#pragma unroll
        for (int nt = 0; nt < 4; ++nt) {
            v16h bfr = frag_b_ptr(&kt[ks * 32][nt * 16 + lr], 72, hi);
            acc[nt] = WMMA(acc[nt], a, bfr);
        }
    }

#pragma unroll
    for (int nt = 0; nt < 4; ++nt)
#pragma unroll
        for (int g = 0; g < 8; ++g) {
            const int a = m0 + g + 8 * hi;               // 0..63 (rows >=49 are pad)
            const int n = n0 + nt * 16 + lr;
            const float v = SCALE * acc[nt][g] + bak[(long)(h * AGENT + min(a, AGENT - 1)) * NQ + n];
            S1[((long)((b * HEADS + h) * 64 + a)) * NQ + n] = (_Float16)v;
        }
}

// ---------------------------------------------------------------------------
// 5) Row softmax over n (3136); S1 has 64-row pitch, P1 compact 49 rows
// ---------------------------------------------------------------------------
__global__ __launch_bounds__(128) void softmax1(const _Float16* __restrict__ S1,
                                                _Float16* __restrict__ P1) {
    const int row = blockIdx.x;                    // 0..(16*8*49-1)
    const int bh = row / AGENT, a = row % AGENT;
    const _Float16* s = S1 + ((long)(bh * 64 + a)) * NQ;
    _Float16* p = P1 + (long)row * NQ;
    __shared__ float red[128];
    const int t = threadIdx.x;

    float m = -3.0e38f;
    for (int i = t; i < NQ; i += 128) m = fmaxf(m, (float)s[i]);
    red[t] = m; __syncthreads();
    for (int o = 64; o > 0; o >>= 1) { if (t < o) red[t] = fmaxf(red[t], red[t + o]); __syncthreads(); }
    m = red[0]; __syncthreads();

    float sum = 0.f;
    for (int i = t; i < NQ; i += 128) sum += __expf((float)s[i] - m);
    red[t] = sum; __syncthreads();
    for (int o = 64; o > 0; o >>= 1) { if (t < o) red[t] += red[t + o]; __syncthreads(); }
    const float inv = 1.0f / red[0];

    for (int i = t; i < NQ; i += 128) p[i] = (_Float16)(__expf((float)s[i] - m) * inv);
}

// ---------------------------------------------------------------------------
// 6) agent_v[b,h,49,64] = P1(49x3136) @ V(3136x64)   per (b,h)
// ---------------------------------------------------------------------------
__global__ __launch_bounds__(128) void agent_v_k(const _Float16* __restrict__ P1,
                                                 const _Float16* __restrict__ kv,
                                                 _Float16* __restrict__ av) {
    const int bh = blockIdx.x;
    const int b = bh >> 3, h = bh & 7;
    const int t = threadIdx.x;
    const int wave = t >> 5, lane = t & 31, lr = lane & 15, hi = lane >> 4;
    const int m0 = wave * 16;
    const int arow = m0 + lr;
    const _Float16* abase = P1 + ((long)(bh * AGENT + min(arow, AGENT - 1))) * NQ;
    const _Float16* vbase = kv + (long)b * NQ * 1024 + 512 + h * DHEAD;

    v8f acc[4];
#pragma unroll
    for (int i = 0; i < 4; ++i) acc[i] = (v8f){0.f, 0.f, 0.f, 0.f, 0.f, 0.f, 0.f, 0.f};

    for (int k0 = 0; k0 < NQ; k0 += 32) {
        v16h a = frag_a_ptr(abase + k0, hi);
        const _Float16* bp = vbase + (long)k0 * 1024 + lr;
#pragma unroll
        for (int nt = 0; nt < 4; ++nt) {
            v16h bfr = frag_b_ptr(bp + nt * 16, 1024, hi);
            acc[nt] = WMMA(acc[nt], a, bfr);
        }
    }

#pragma unroll
    for (int nt = 0; nt < 4; ++nt)
#pragma unroll
        for (int g = 0; g < 8; ++g) {
            const int a = m0 + g + 8 * hi;
            if (a < AGENT)
                av[((long)(bh * AGENT + a)) * DHEAD + nt * 16 + lr] = (_Float16)acc[nt][g];
        }
}

// ---------------------------------------------------------------------------
// 7) Fused stage-2: S2 = scale*Q@agent^T + bias -> softmax(49) -> @agent_v -> T (f16)
// ---------------------------------------------------------------------------
__global__ __launch_bounds__(128) void stage2(const _Float16* __restrict__ q16,
                                              const _Float16* __restrict__ ag,
                                              const float* __restrict__ bqa,
                                              const _Float16* __restrict__ av,
                                              _Float16* __restrict__ T) {
    const int b = blockIdx.z, h = blockIdx.y;
    const int q0 = blockIdx.x * 64;
    __shared__ _Float16 agT[64][72];  // [d][a]
    __shared__ _Float16 p2[64][72];   // [row][a]
    const int t = threadIdx.x;
    for (int e = t; e < 64 * 64; e += 128) {
        const int a = e >> 6, d = e & 63;
        agT[d][a] = (a < AGENT) ? ag[((long)(b * AGENT + a)) * CQ + h * DHEAD + d]
                                : (_Float16)0.0f;
    }
    __syncthreads();

    const int wave = t >> 5, lane = t & 31, lr = lane & 15, hi = lane >> 4;
    const int m0 = wave * 16;
    const _Float16* qbase = q16 + ((long)(b * NQ + q0 + m0 + lr)) * CQ + h * DHEAD;

    v8f acc[4];
#pragma unroll
    for (int i = 0; i < 4; ++i) acc[i] = (v8f){0.f, 0.f, 0.f, 0.f, 0.f, 0.f, 0.f, 0.f};

#pragma unroll
    for (int ks = 0; ks < 2; ++ks) {
        v16h a = frag_a_ptr(qbase + ks * 32, hi);
#pragma unroll
        for (int nt = 0; nt < 4; ++nt) {
            v16h bfr = frag_b_ptr(&agT[ks * 32][nt * 16 + lr], 72, hi);
            acc[nt] = WMMA(acc[nt], a, bfr);
        }
    }

#pragma unroll
    for (int g = 0; g < 8; ++g) {
        const int r = m0 + g + 8 * hi;
        const int n = q0 + r;
        float v[4];
#pragma unroll
        for (int nt = 0; nt < 4; ++nt) {
            const int a = nt * 16 + lr;
            v[nt] = (a < AGENT) ? SCALE * acc[nt][g] + bqa[((long)(h * NQ + n)) * 64 + a]
                                : -3.0e38f;
        }
        float mx = fmaxf(fmaxf(v[0], v[1]), fmaxf(v[2], v[3]));
        mx = fmaxf(mx, __shfl_xor(mx, 1));
        mx = fmaxf(mx, __shfl_xor(mx, 2));
        mx = fmaxf(mx, __shfl_xor(mx, 4));
        mx = fmaxf(mx, __shfl_xor(mx, 8));
        float e4[4], s = 0.f;
#pragma unroll
        for (int nt = 0; nt < 4; ++nt) { e4[nt] = __expf(v[nt] - mx); s += e4[nt]; }
        s += __shfl_xor(s, 1);
        s += __shfl_xor(s, 2);
        s += __shfl_xor(s, 4);
        s += __shfl_xor(s, 8);
        const float inv = 1.0f / s;
#pragma unroll
        for (int nt = 0; nt < 4; ++nt) p2[r][nt * 16 + lr] = (_Float16)(e4[nt] * inv);
    }
    __syncthreads();

    v8f acc2[4];
#pragma unroll
    for (int i = 0; i < 4; ++i) acc2[i] = (v8f){0.f, 0.f, 0.f, 0.f, 0.f, 0.f, 0.f, 0.f};

#pragma unroll
    for (int ks = 0; ks < 2; ++ks) {
        v16h a = frag_a_ptr(&p2[m0 + lr][ks * 32], hi);
        const _Float16* bp = av + ((long)((b * HEADS + h) * AGENT)) * DHEAD + lr;
#pragma unroll
        for (int nt = 0; nt < 4; ++nt) {
            v16h bfr;
#pragma unroll
            for (int j = 0; j < 16; ++j) {
                const int kr = ks * 32 + (j & 7) + 8 * hi + 16 * (j >> 3);
                bfr[j] = (kr < AGENT) ? bp[kr * DHEAD + nt * 16] : (_Float16)0.0f;
            }
            acc2[nt] = WMMA(acc2[nt], a, bfr);
        }
    }

#pragma unroll
    for (int nt = 0; nt < 4; ++nt)
#pragma unroll
        for (int g = 0; g < 8; ++g) {
            const int r = m0 + g + 8 * hi;
            const int n = q0 + r;
            T[((long)(b * NQ + n)) * CQ + h * DHEAD + nt * 16 + lr] = (_Float16)acc2[nt][g];
        }
}

// ---------------------------------------------------------------------------
// 8) Depthwise 3x3 conv on v, accumulated into T (f16)
// ---------------------------------------------------------------------------
__global__ __launch_bounds__(256) void dwc_add(const _Float16* __restrict__ kv,
                                               const float* __restrict__ w,
                                               const float* __restrict__ bias,
                                               _Float16* __restrict__ T) {
    const long idx = (long)blockIdx.x * 256 + threadIdx.x;
    if (idx >= (long)MROWS * CQ) return;
    const int c = (int)(idx & (CQ - 1));
    const long bn = idx >> 9;
    const int n = (int)(bn % NQ);
    const int b = (int)(bn / NQ);
    const int y = n / HH, x = n % HH;
    float acc = bias[c];
#pragma unroll
    for (int dy = 0; dy < 3; ++dy) {
        const int yy = y + dy - 1;
        if (yy < 0 || yy >= HH) continue;
#pragma unroll
        for (int dx = 0; dx < 3; ++dx) {
            const int xx = x + dx - 1;
            if (xx < 0 || xx >= HH) continue;
            acc += (float)kv[((long)(b * NQ + yy * HH + xx)) * 1024 + 512 + c] *
                   w[c * 9 + dy * 3 + dx];
        }
    }
    T[idx] = (_Float16)((float)T[idx] + acc);
}

// ---------------------------------------------------------------------------
// launch
// ---------------------------------------------------------------------------
extern "C" void kernel_launch(void* const* d_in, const int* in_sizes, int n_in,
                              void* d_out, int out_size, void* d_ws, size_t ws_size,
                              hipStream_t stream) {
    (void)in_sizes; (void)n_in; (void)out_size; (void)ws_size;
    const float* x      = (const float*)d_in[0];
    const float* q_w    = (const float*)d_in[1];
    const float* kv_w   = (const float*)d_in[2];
    const float* proj_w = (const float*)d_in[3];
    const float* proj_b = (const float*)d_in[4];
    const float* dwc_w  = (const float*)d_in[5];
    const float* dwc_b  = (const float*)d_in[6];
    const float* an_b   = (const float*)d_in[7];
    const float* na_b   = (const float*)d_in[8];
    const float* ah_b   = (const float*)d_in[9];
    const float* aw_b   = (const float*)d_in[10];
    const float* ha_b   = (const float*)d_in[11];
    const float* wa_b   = (const float*)d_in[12];
    float* out = (float*)d_out;

    // workspace layout (bytes, 256-aligned)
    char* ws = (char*)d_ws;
    size_t off = 0;
    auto alloc = [&](size_t bytes) { char* p = ws + off; off = (off + bytes + 255) & ~(size_t)255; return p; };
    const size_t sMC  = (size_t)MROWS * CQ * 2;        // f16 M x 512
    _Float16* x16  = (_Float16*)alloc(sMC);            // reused for padded S1 (same size)
    _Float16* q16  = (_Float16*)alloc(sMC);
    _Float16* kv16 = (_Float16*)alloc((size_t)MROWS * 1024 * 2);
    _Float16* T16  = (_Float16*)alloc(sMC);
    _Float16* P1   = (_Float16*)alloc((size_t)BQ * HEADS * AGENT * NQ * 2);
    _Float16* ag16 = (_Float16*)alloc((size_t)BQ * AGENT * CQ * 2);
    _Float16* av16 = (_Float16*)alloc((size_t)BQ * HEADS * AGENT * DHEAD * 2);
    float* bak     = (float*)alloc((size_t)HEADS * AGENT * NQ * 4);
    float* bqa     = (float*)alloc((size_t)HEADS * NQ * 64 * 4);
    _Float16* wq16 = (_Float16*)alloc((size_t)CQ * CQ * 2);
    _Float16* wkv16= (_Float16*)alloc((size_t)CQ * 2 * CQ * 2);
    _Float16* wp16 = (_Float16*)alloc((size_t)CQ * CQ * 2);
    _Float16* S1   = x16;   // alias: x16 dead after the q/kv GEMMs; 16*8*64*3136*2 == sMC

    const long nx = (long)MROWS * CQ;

    // converts: x and the three weight matrices -> f16
    cvt_f32_f16<<<(int)((nx + 255) / 256), 256, 0, stream>>>(x, x16, nx);
    cvt_f32_f16<<<(CQ * CQ + 255) / 256, 256, 0, stream>>>(q_w, wq16, (long)CQ * CQ);
    cvt_f32_f16<<<(CQ * 2 * CQ + 255) / 256, 256, 0, stream>>>(kv_w, wkv16, (long)CQ * 2 * CQ);
    cvt_f32_f16<<<(CQ * CQ + 255) / 256, 256, 0, stream>>>(proj_w, wp16, (long)CQ * CQ);

    // q = x @ q_w (f16 out)
    gemm16<512, 512, false, true><<<dim3(512 / 256, MROWS / 128), 256, 0, stream>>>(
        x16, wq16, nullptr, nullptr, q16);
    // kv = x @ kv_w (f16 out, K|V interleaved, ld 1024)
    gemm16<1024, 512, false, true><<<dim3(1024 / 256, MROWS / 128), 256, 0, stream>>>(
        x16, wkv16, nullptr, nullptr, kv16);
    // agent pooling
    pool_agent<<<dim3(AGENT, BQ), 256, 0, stream>>>(q16, ag16);
    // bias tables
    bias_build<<<(HEADS * AGENT * NQ + 255) / 256, 256, 0, stream>>>(an_b, na_b, ah_b, aw_b,
                                                                     ha_b, wa_b, bak, bqa);
    // stage-1 scores + bias (padded 64-row pitch)
    s1_scores<<<dim3(NQ / 64, HEADS, BQ), 128, 0, stream>>>(ag16, kv16, bak, S1);
    // softmax over n
    softmax1<<<BQ * HEADS * AGENT, 128, 0, stream>>>(S1, P1);
    // agent_v
    agent_v_k<<<BQ * HEADS, 128, 0, stream>>>(P1, kv16, av16);
    // fused stage-2 -> T
    stage2<<<dim3(NQ / 64, HEADS, BQ), 128, 0, stream>>>(q16, ag16, bqa, av16, T16);
    // T += dwconv(v)
    dwc_add<<<(int)((nx + 255) / 256), 256, 0, stream>>>(kv16, dwc_w, dwc_b, T16);
    // out = T @ proj_w + proj_b (f32 out)
    gemm16<512, 512, true, false><<<dim3(512 / 256, MROWS / 128), 256, 0, stream>>>(
        T16, wp16, proj_b, out, nullptr);
}
